// CoOccurrenceModule_11098195493225
// MI455X (gfx1250) — matrix operand order repaired and checked by reading, not compiled, not run on hardware
//
#include <hip/hip_runtime.h>

// CDNA5 / gfx1250, wave32.
typedef __attribute__((ext_vector_type(8)))  _Float16 v8h;
typedef __attribute__((ext_vector_type(16))) _Float16 v16h;
typedef __attribute__((ext_vector_type(8)))  float    v8f;

#define Hn   112
#define Wn   112
#define Cc   64
#define LCOL 128          // padded LDS columns (covers w = -2 .. 125, zeroed)
#define PSTR 72           // halves per pixel (64 ch + 8 pad) -> 144B stride, 16B aligned
#define NPIX (5 * LCOL)   // 640 staged pixels
#define XS_HALVES (NPIX * PSTR)                 // 46080 halves = 92160 B
#define OFF_NORM  (XS_HALVES * 2)               // 92160
#define OFF_DOTS  (OFF_NORM + NPIX * 4)         // + 2560
#define OFF_SW    (OFF_DOTS + 25 * Wn * 4)      // + 11200
#define SMEM_BYTES (OFF_SW + 32 * 4)            // ~106 KB dynamic LDS
#define DUMP_SLOT 28      // swl[28]: scratch sink for out-of-band C elements

union V16 { v16h v; v8h h[2]; };

__global__ __launch_bounds__(512)
void cooc_kernel(const float* __restrict__ x, const float* __restrict__ sw,
                 float* __restrict__ out) {
    extern __shared__ unsigned char smem[];
    _Float16* xs    = (_Float16*)smem;
    float*    normv = (float*)(smem + OFF_NORM);
    float*    dots  = (float*)(smem + OFF_DOTS);
    float*    swl   = (float*)(smem + OFF_SW);

    const int tid = threadIdx.x;
    const int bb  = blockIdx.x / Hn;   // batch
    const int h0  = blockIdx.x % Hn;   // output row

    if (tid < 25) swl[tid] = sw[tid];

    // ---- Phase 1: stage x[b, :, h0-2..h0+2, -2..125] -> LDS f16 (zero-padded).
    // Order (row, ch, col): col is lane-minor -> coalesced 4B global loads.
    for (int i = tid; i < 5 * Cc * LCOL; i += blockDim.x) {
        int col = i & (LCOL - 1);
        int ch  = (i >> 7) & (Cc - 1);
        int row = i >> 13;
        int w   = col - 2;
        int hh  = h0 + row - 2;
        float v = 0.0f;
        if (w >= 0 && w < Wn && hh >= 0 && hh < Hn)
            v = x[(((size_t)bb * Cc + ch) * Hn + hh) * Wn + w];
        xs[(size_t)(row * LCOL + col) * PSTR + ch] = (_Float16)v;
    }
    __syncthreads();

    // ---- Phase 1b: per-pixel L2 norms from the same f16 data (keeps sim <= 1).
    for (int t = tid; t < NPIX; t += blockDim.x) {
        const _Float16* p = xs + (size_t)t * PSTR;
        float s = 0.0f;
        #pragma unroll
        for (int c = 0; c < Cc; ++c) { float f = (float)p[c]; s += f * f; }
        normv[t] = sqrtf(s);
    }
    __syncthreads();

    // ---- Phase 2: Gram bands via v_wmma_f32_16x16x32_f16.
    // Task = (16-pixel block, neighbor tile): 7*2 = 14 tasks over 16 waves.
    // A fragment (center pixels) loaded once per task, reused across 5 row offsets.
    {
        const int wid  = tid >> 5;
        const int lane = tid & 31;
        const int ln   = lane & 15;
        const int sel  = lane >> 4;
        for (int t = wid; t < 14; t += 16) {
            const int blk = t >> 1, nt = t & 1;
            const int w0  = blk * 16;
            // A fragment: 16 center pixels (row 2), K = channels.
            // lanes 0-15: K 0-7 & 16-23 ; lanes 16-31: K 8-15 & 24-31 (per half).
            const _Float16* ap = xs + (size_t)(2 * LCOL + w0 + 2 + ln) * PSTR;
            V16 a0, a1;
            a0.h[0] = *(const v8h*)(ap +      8 * sel);
            a0.h[1] = *(const v8h*)(ap + 16 + 8 * sel);
            a1.h[0] = *(const v8h*)(ap + 32 + 8 * sel);
            a1.h[1] = *(const v8h*)(ap + 48 + 8 * sel);
            for (int di = 0; di < 5; ++di) {
                // B fragment: neighbor pixels; lanes 0-15: K 0-7 & 16-23 ; lanes 16-31: K 8-15 & 24-31.
                const _Float16* bp =
                    xs + (size_t)(di * LCOL + w0 + 16 * nt + ln) * PSTR + 8 * sel;
                V16 b0, b1;
                b0.h[0] = *(const v8h*)(bp + 0);
                b0.h[1] = *(const v8h*)(bp + 16);
                b1.h[0] = *(const v8h*)(bp + 32);
                b1.h[1] = *(const v8h*)(bp + 48);
                v8f c = {};
                c = __builtin_amdgcn_wmma_f32_16x16x32_f16(
                        false, a0.v, false, b0.v, (short)0, c, false, false);
                c = __builtin_amdgcn_wmma_f32_16x16x32_f16(
                        false, a1.v, false, b1.v, (short)0, c, false, false);
                // Scatter the +/-2 band: C[mm][n] -> dots[k][w0+mm].
                // Branch-free: out-of-band elements go to an LDS dump slot
                // (address select instead of EXEC-masked store).
                #pragma unroll
                for (int r = 0; r < 8; ++r) {
                    int mm  = r + 8 * sel;
                    int dj2 = 16 * nt + ln - mm;          // dj + 2
                    bool ok = (unsigned)dj2 <= 4u;
                    float* dst = ok ? &dots[(di * 5 + dj2) * Wn + (w0 + mm)]
                                    : &swl[DUMP_SLOT];
                    *dst = c[r];
                }
            }
        }
    }
    __syncthreads();

    // ---- Phase 3: dots -> weights: (cos_sim + 1)/2 * spatial_weight[k].
    for (int t = tid; t < 25 * Wn; t += blockDim.x) {
        int k = t / Wn, w = t % Wn;
        int di = k / 5, j = k % 5;
        float d  = dots[t];
        float nx = normv[2 * LCOL + w + 2];
        float np = normv[di * LCOL + w + j];
        float sim = d / fmaxf(nx * np, 1e-8f);
        dots[t] = (sim + 1.0f) * 0.5f * swl[k];
    }
    __syncthreads();

    // ---- Phase 4: out[c,w] = sum_k weight[k,w] * patch[c,k,w].
    for (int t = tid; t < (Cc / 8) * Wn; t += blockDim.x) {
        int w  = t % Wn;          // lane-minor -> coalesced stores
        int cg = t / Wn;          // 8-channel group
        float acc[8] = {0, 0, 0, 0, 0, 0, 0, 0};
        #pragma unroll
        for (int k = 0; k < 25; ++k) {
            int di = k / 5, j = k % 5;
            float wt = dots[k * Wn + w];
            v8h xv = *(const v8h*)(xs + (size_t)(di * LCOL + w + j) * PSTR + cg * 8);
            #pragma unroll
            for (int e = 0; e < 8; ++e) acc[e] += wt * (float)xv[e];
        }
        int chb = cg * 8;
        #pragma unroll
        for (int e = 0; e < 8; ++e)
            out[(((size_t)bb * Cc + chb + e) * Hn + h0) * Wn + w] = acc[e];
    }
}

extern "C" void kernel_launch(void* const* d_in, const int* in_sizes, int n_in,
                              void* d_out, int out_size, void* d_ws, size_t ws_size,
                              hipStream_t stream) {
    const float* x  = (const float*)d_in[0];   // (8, 64, 112, 112) f32
    const float* sw = (const float*)d_in[1];   // (25,) f32
    float* out = (float*)d_out;                // (8, 64, 112, 112) f32
    dim3 grid(8 * Hn);                         // one workgroup per (batch, row)
    dim3 block(512);                           // 16 wave32 waves
    cooc_kernel<<<grid, block, SMEM_BYTES, stream>>>(x, sw, out);
}
